// BatchedVectorAttention_13460427505894
// MI455X (gfx1250) — matrix operand order repaired and checked by reading, not compiled, not run on hardware
//
#include <hip/hip_runtime.h>
#include <hip/hip_bf16.h>
#include <math.h>

// ---------------------------------------------------------------------------
// Problem constants (from reference): N=8, L=2048, C=512, H=512, C_HALF=256
// ---------------------------------------------------------------------------
#define NB   8
#define LS   2048
#define CC   512
#define HH   512
#define CHF  256
#define NLR  (NB * LS)   // 16384 total rows

typedef __attribute__((ext_vector_type(16))) _Float16 v16h;
typedef __attribute__((ext_vector_type(8)))  _Float16 v8h;
typedef __attribute__((ext_vector_type(8)))  float    v8f;

__device__ __forceinline__ v16h cat8(v8h lo, v8h hi) {
    return __builtin_shufflevector(lo, hi, 0,1,2,3,4,5,6,7,8,9,10,11,12,13,14,15);
}

__device__ __forceinline__ v8f wmma_f16(v16h a, v16h b, v8f c) {
    // D = A(16x32 f16) * B(32x16 f16) + C(16x16 f32)
    return __builtin_amdgcn_wmma_f32_16x16x32_f16(
        /*neg_a=*/false, a, /*neg_b=*/false, b,
        /*c_mod=*/(short)0, c, /*reuse_a=*/false, /*reuse_b=*/false);
}

// ---------------------------------------------------------------------------
// Prep kernels: f32 -> f16 convert, and f32 W[K][N] -> f16 Wt[N][K] transpose
// ---------------------------------------------------------------------------
__global__ void convert_f16_kernel(const float* __restrict__ in,
                                   _Float16* __restrict__ out, int n) {
    int i = blockIdx.x * blockDim.x + threadIdx.x;
    if (i < n) out[i] = (_Float16)in[i];
}

__global__ void transpose_f16_kernel(const float* __restrict__ W,
                                     _Float16* __restrict__ Wt, int K, int Nn) {
    int i = blockIdx.x * blockDim.x + threadIdx.x;
    if (i >= K * Nn) return;
    int k = i / Nn;
    int n = i - k * Nn;
    Wt[(size_t)n * K + k] = (_Float16)W[(size_t)k * Nn + n];
}

// ---------------------------------------------------------------------------
// Generic WMMA GEMM:  D[row][col] = sum_k A[row][k] * Wt[col][k]  + bias[col]
//   A:  f16 row-major [Mrows x K]
//   Wt: f16 "B-friendly" [Nout x K] (i.e. original W transposed)
// Block = 128 threads = 4 waves; block tile = 16 rows x 256 cols
// (wave w -> 64 cols = 4 WMMA C tiles). B fragments for a k-step are
// preloaded as a group so loads clause together and WMMAs issue back-to-back.
// Epilogues:
//   0: +bias, store f16 row-major
//   1: +bias, store f16 TRANSPOSED per batch: out[b][col][l]  (for psi)
//   2: +bias, LeakyReLU(0.2), store f16 row-major
//   3: +bias, tanh, + residual(f32), store f32 row-major (final output)
// ---------------------------------------------------------------------------
#define EPI_F16        0
#define EPI_F16_TRANS  1
#define EPI_LEAKY      2
#define EPI_TANH_RES   3

template <int EPI, int K, int NOUT>
__global__ __launch_bounds__(128)
void gemm_wmma_kernel(const _Float16* __restrict__ A,
                      const _Float16* __restrict__ Wt,
                      const float* __restrict__ bias,
                      _Float16* __restrict__ outh,
                      float* __restrict__ outf,
                      const float* __restrict__ resid) {
    const int lane = threadIdx.x & 31;
    const int wv   = threadIdx.x >> 5;
    const int m    = lane & 15;     // A: row-in-tile / B,C: col-in-tile
    const int g    = lane >> 4;     // half-select
    const int row0 = blockIdx.x * 16;
    const int col0 = blockIdx.y * 256 + wv * 64;

    v8f acc[4] = {};

    const _Float16* arow = A  + (size_t)(row0 + m) * K + 16 * g;
    const _Float16* bcol = Wt + (size_t)(col0 + m) * K + 16 * g;

#pragma unroll 4
    for (int kk = 0; kk < K; kk += 32) {
        __builtin_prefetch(arow + kk + 128, 0, 1);   // global_prefetch_b8
        // batch all loads for this k-step, then run the WMMA chain
        v8h alo = *(const v8h*)(arow + kk);
        v8h ahi = *(const v8h*)(arow + kk + 8);
        v16h bbs[4];
#pragma unroll
        for (int t = 0; t < 4; ++t)
            bbs[t] = *(const v16h*)(bcol + (size_t)t * 16 * K + kk);
        v16h a = cat8(alo, ahi);
#pragma unroll
        for (int t = 0; t < 4; ++t)
            acc[t] = wmma_f16(a, bbs[t], acc[t]);
    }

#pragma unroll
    for (int t = 0; t < 4; ++t) {
        const int col = col0 + t * 16 + m;
        const float bv = bias[col];
#pragma unroll
        for (int r = 0; r < 8; ++r) {
            const int grow = row0 + r + 8 * g;   // global row (M = r + 8*g)
            float v = acc[t][r] + bv;
            if (EPI == EPI_LEAKY) v = v > 0.f ? v : 0.2f * v;
            if (EPI == EPI_F16 || EPI == EPI_LEAKY) {
                outh[(size_t)grow * NOUT + col] = (_Float16)v;
            } else if (EPI == EPI_F16_TRANS) {
                const int b = grow / LS;
                const int l = grow - b * LS;
                outh[((size_t)b * NOUT + col) * LS + l] = (_Float16)v;
            } else { // EPI_TANH_RES
                float o = tanhf(v) + resid[(size_t)grow * NOUT + col];
                outf[(size_t)grow * NOUT + col] = o;
            }
        }
    }
}

// ---------------------------------------------------------------------------
// Flash-attention kernel (online softmax, no LxL materialization).
// Grid: (L/32, N). Block: 256 threads = 8 waves.
// Each block: 32 query rows of batch b, full H=512 output columns.
//   wave w: row-half rh = w>>2 (16 rows), col-slice cw = w&3.
// Block's phi rows (32x512 f16 = 32KB) staged in LDS once -- via
// GLOBAL_LOAD_ASYNC_TO_LDS_B128 (ASYNCcnt path) when the toolchain exposes
// the builtin, else a plain v8h copy. A-fragments then come from
// ds_load_b128 for every key chunk instead of re-reading L2.
// Key chunks of 64:
//   S(32x64) = phi_rows . theta_chunk^T   (wave (rh,cw) -> 16x16 tile)
//   -> LDS; 32 row-threads do online softmax, write P=exp(S-m) as f16 to LDS
//   acc = acc*scale + P . psi_chunk       (psi stored transposed [b][h][l])
// ---------------------------------------------------------------------------
__global__ __launch_bounds__(256)
void attn_flash_kernel(const _Float16* __restrict__ phi,
                       const _Float16* __restrict__ theta,
                       const _Float16* __restrict__ psiT,
                       _Float16* __restrict__ outh) {
    __shared__ __attribute__((aligned(32))) _Float16 phiS[32][HH];  // 32 KB
    __shared__ __attribute__((aligned(32))) float    Sbuf[32][64];  //  8 KB
    __shared__ __attribute__((aligned(32))) _Float16 Pbuf[32][64];  //  4 KB
    __shared__ float rowmax[32], rowsum[32], scaleb[32];

    const int tid  = threadIdx.x;
    const int lane = tid & 31;
    const int wv   = tid >> 5;
    const int rh   = wv >> 2;       // row-half: 0 or 1
    const int cw   = wv & 3;        // col-slice: 0..3
    const int m    = lane & 15;
    const int g    = lane >> 4;
    const int b    = blockIdx.y;
    const int row0 = blockIdx.x * 32;

    // ---- stage this block's phi rows into LDS (async DMA when available) ----
    {
        const _Float16* src = phi + ((size_t)b * LS + row0) * HH;
#if __has_builtin(__builtin_amdgcn_global_load_async_to_lds_b128)
        // param types per compiler diagnostic: (v4i AS1*, v4i AS3*, Ii, Ii)
        typedef int v4i __attribute__((vector_size(16)));
        typedef __attribute__((address_space(1))) v4i g_v4i;
        typedef __attribute__((address_space(3))) v4i l_v4i;
        for (int i = tid; i < 32 * HH / 8; i += 256) {
            const int r = i >> 6;            // HH/8 = 64 chunks per row
            const int c = (i & 63) * 8;
            __builtin_amdgcn_global_load_async_to_lds_b128(
                (g_v4i*)(src + (size_t)r * HH + c),
                (l_v4i*)(&phiS[r][c]), /*offset=*/0, /*cpol=*/0);
        }
#if __has_builtin(__builtin_amdgcn_s_wait_asynccnt)
        __builtin_amdgcn_s_wait_asynccnt(0);
#else
        asm volatile("s_wait_asynccnt 0x0" ::: "memory");
#endif
#else
        for (int i = tid; i < 32 * HH / 8; i += 256) {
            const int r = i >> 6;            // HH/8 = 64 chunks per row
            const int c = (i & 63) * 8;
            *(v8h*)(&phiS[r][c]) = *(const v8h*)(src + (size_t)r * HH + c);
        }
#endif
    }
    if (tid < 32) { rowmax[tid] = -1e30f; rowsum[tid] = 0.f; }
    __syncthreads();

    v8f acc[8] = {};   // 16 rows x 128 cols per wave, f32

    for (int j = 0; j < LS; j += 64) {
        // ---- S tile: wave (rh,cw): rows [rh*16,+16), keys [j+cw*16,+16) ----
        v8f s = {};
        const _Float16* thRow =
            theta + ((size_t)b * LS + j + cw * 16 + m) * HH + 16 * g;
#pragma unroll 4
        for (int kk = 0; kk < HH; kk += 32) {
            v8h alo = *(const v8h*)(&phiS[rh * 16 + m][kk + 16 * g]);
            v8h ahi = *(const v8h*)(&phiS[rh * 16 + m][kk + 16 * g + 8]);
            v16h bb = *(const v16h*)(thRow + kk);
            s = wmma_f16(cat8(alo, ahi), bb, s);
        }
#pragma unroll
        for (int r = 0; r < 8; ++r) Sbuf[rh * 16 + r + 8 * g][cw * 16 + m] = s[r];
        __syncthreads();

        // ---- online softmax update: one thread per query row ----
        if (tid < 32) {
            float mx = rowmax[tid];
            float cm = mx;
            for (int k = 0; k < 64; ++k) cm = fmaxf(cm, Sbuf[tid][k]);
            float sc = __expf(mx - cm);     // rescale factor for old state
            float rs = 0.f;
            for (int k = 0; k < 64; ++k) {
                float p = __expf(Sbuf[tid][k] - cm);
                Pbuf[tid][k] = (_Float16)p;
                rs += p;
            }
            rowsum[tid] = rowsum[tid] * sc + rs;
            rowmax[tid] = cm;
            scaleb[tid] = sc;
        }
        __syncthreads();

        // ---- rescale accumulators ----
        float fr[8];
#pragma unroll
        for (int r = 0; r < 8; ++r) fr[r] = scaleb[rh * 16 + r + 8 * g];
#pragma unroll
        for (int t = 0; t < 8; ++t)
#pragma unroll
            for (int r = 0; r < 8; ++r) acc[t][r] *= fr[r];

        // ---- acc += P(16x64) . psi_chunk(64 x 128 cols per wave) ----
        const _Float16* psiBase =
            psiT + ((size_t)b * HH + cw * 128 + m) * LS + j + 16 * g;
#pragma unroll
        for (int k0 = 0; k0 < 64; k0 += 32) {
            v8h plo = *(const v8h*)(&Pbuf[rh * 16 + m][k0 + 16 * g]);
            v8h phi2 = *(const v8h*)(&Pbuf[rh * 16 + m][k0 + 16 * g + 8]);
            // batch the 8 B-fragment loads, then back-to-back WMMAs
            v16h bbs[8];
#pragma unroll
            for (int t = 0; t < 8; ++t)
                bbs[t] = *(const v16h*)(psiBase + (size_t)t * 16 * LS + k0);
            v16h a = cat8(plo, phi2);
#pragma unroll
            for (int t = 0; t < 8; ++t)
                acc[t] = wmma_f16(a, bbs[t], acc[t]);
        }
        // next iteration's Sbuf write is separated from this Pbuf read by the
        // pre-softmax __syncthreads(); Pbuf writes happen only after it.
    }
    __syncthreads();

    // ---- epilogue: divide by softmax denominator, store f16 ----
    float li[8];
#pragma unroll
    for (int r = 0; r < 8; ++r) li[r] = 1.0f / rowsum[rh * 16 + r + 8 * g];
#pragma unroll
    for (int t = 0; t < 8; ++t) {
        const int col = cw * 128 + t * 16 + m;
#pragma unroll
        for (int r = 0; r < 8; ++r) {
            outh[((size_t)b * LS + row0 + rh * 16 + r + 8 * g) * HH + col] =
                (_Float16)(acc[t][r] * li[r]);
        }
    }
}

// ---------------------------------------------------------------------------
// Host-side launcher
// ---------------------------------------------------------------------------
extern "C" void kernel_launch(void* const* d_in, const int* in_sizes, int n_in,
                              void* d_out, int out_size, void* d_ws, size_t ws_size,
                              hipStream_t stream) {
    (void)in_sizes; (void)n_in; (void)out_size; (void)ws_size;

    const float* x       = (const float*)d_in[0];
    const float* theta_w = (const float*)d_in[1];
    const float* theta_b = (const float*)d_in[2];
    const float* phi_w   = (const float*)d_in[3];
    const float* phi_b   = (const float*)d_in[4];
    const float* psi_w   = (const float*)d_in[5];
    const float* psi_b   = (const float*)d_in[6];
    const float* r1_w    = (const float*)d_in[7];
    const float* r1_b    = (const float*)d_in[8];
    const float* r2_w    = (const float*)d_in[9];
    const float* r2_b    = (const float*)d_in[10];

    // carve f16 workspace (256B-aligned chunks)
    char* p = (char*)d_ws;
    auto carve = [&](size_t elems) {
        _Float16* r = (_Float16*)p;
        p += (elems * sizeof(_Float16) + 255) & ~(size_t)255;
        return r;
    };
    _Float16* xh      = carve((size_t)NLR * CC);   // x as f16
    _Float16* thetaWt = carve((size_t)HH * CC);    // theta_w^T [H][C]
    _Float16* phiWt   = carve((size_t)HH * CC);
    _Float16* psiWt   = carve((size_t)HH * CC);
    _Float16* r1Wt    = carve((size_t)CHF * HH);   // r1_w^T [256][512]
    _Float16* r2Wt    = carve((size_t)CC * CHF);   // r2_w^T [512][256]
    _Float16* theta16 = carve((size_t)NLR * HH);   // [n*l][h]
    _Float16* phi16   = carve((size_t)NLR * HH);   // [n*l][h]
    _Float16* psiT    = carve((size_t)NB * HH * LS); // [b][h][l]
    _Float16* xaddh   = carve((size_t)NLR * HH);   // attention out, f16
    _Float16* h1      = carve((size_t)NLR * CHF);  // MLP hidden, f16

    const dim3 blk(128);

    // ---- prep ----
    convert_f16_kernel<<<(NLR * CC + 255) / 256, 256, 0, stream>>>(x, xh, NLR * CC);
    transpose_f16_kernel<<<(CC * HH + 255) / 256, 256, 0, stream>>>(theta_w, thetaWt, CC, HH);
    transpose_f16_kernel<<<(CC * HH + 255) / 256, 256, 0, stream>>>(phi_w,   phiWt,   CC, HH);
    transpose_f16_kernel<<<(CC * HH + 255) / 256, 256, 0, stream>>>(psi_w,   psiWt,   CC, HH);
    transpose_f16_kernel<<<(HH * CHF + 255) / 256, 256, 0, stream>>>(r1_w,   r1Wt,    HH, CHF);
    transpose_f16_kernel<<<(CHF * CC + 255) / 256, 256, 0, stream>>>(r2_w,   r2Wt,    CHF, CC);

    // ---- projections (WMMA GEMMs) ----
    gemm_wmma_kernel<EPI_F16, CC, HH><<<dim3(NLR / 16, HH / 256), blk, 0, stream>>>(
        xh, thetaWt, theta_b, theta16, nullptr, nullptr);
    gemm_wmma_kernel<EPI_F16, CC, HH><<<dim3(NLR / 16, HH / 256), blk, 0, stream>>>(
        xh, phiWt, phi_b, phi16, nullptr, nullptr);
    gemm_wmma_kernel<EPI_F16_TRANS, CC, HH><<<dim3(NLR / 16, HH / 256), blk, 0, stream>>>(
        xh, psiWt, psi_b, psiT, nullptr, nullptr);

    // ---- fused flash attention ----
    attn_flash_kernel<<<dim3(LS / 32, NB), dim3(256), 0, stream>>>(
        phi16, theta16, psiT, xaddh);

    // ---- recover MLP + residual ----
    gemm_wmma_kernel<EPI_LEAKY, HH, CHF><<<dim3(NLR / 16, CHF / 256), blk, 0, stream>>>(
        xaddh, r1Wt, r1_b, h1, nullptr, nullptr);
    gemm_wmma_kernel<EPI_TANH_RES, CHF, CC><<<dim3(NLR / 16, CC / 256), blk, 0, stream>>>(
        h1, r2Wt, r2_b, nullptr, (float*)d_out, x);
}
